// MixtureOfDepths_89421219103400
// MI455X (gfx1250) — compile-verified, matrix operand order checked by reading, not run on hardware
//
#include <hip/hip_runtime.h>
#include <hip/hip_bf16.h>
#include <math.h>

// ---------------------------------------------------------------------------
// Mixture-of-Depths layer for MI455X (gfx1250, wave32, WMMA bf16 16x16x32,
// Tensor Data Mover double-buffered tile staging).
//   B=4, S=4096 -> 16384 tokens; D=2048, D_CONF=512, D_FF=8192.
//
// Workspace layout (assumed ws_size >= ~384 MB):
//   off 0        : H      bf16 [NTOK][D_FF]      256 MB (GELU intermediate)
//   +256 MB      : Xbf    bf16 [NTOK][D_MODEL]    64 MB
//   +320 MB      : Wf1T   bf16 [D_FF][D_MODEL]    32 MB (transposed)
//   +352 MB      : Wf2T   bf16 [D_MODEL][D_FF]    32 MB (transposed)
//   +384 MB      : mask   u32  [NTOK]             64 KB
// ---------------------------------------------------------------------------

#define D_MODEL   2048
#define D_CONF    512
#define D_FF      8192
#define NTOK      16384
#define THRESHOLD 0.8f
#define MIN_LAYERS 1

typedef __attribute__((ext_vector_type(16))) __bf16 v16bf;
typedef __attribute__((ext_vector_type(8)))  float  v8f;
typedef __attribute__((ext_vector_type(4)))  unsigned int v4u;
typedef __attribute__((ext_vector_type(8)))  int    v8i;
typedef __attribute__((ext_vector_type(4)))  int    v4i;

union Frag16 { v16bf v; unsigned u[8]; };

__device__ __forceinline__ unsigned short f2bf_bits(float x) {
  unsigned u = __builtin_bit_cast(unsigned, x);
  unsigned r = u + 0x7FFFu + ((u >> 16) & 1u);   // round-to-nearest-even
  return (unsigned short)(r >> 16);
}

__device__ __forceinline__ float gelu_exact(float x) {
  return 0.5f * x * (1.0f + erff(x * 0.70710678118654752f));
}

// Load a 16x32 bf16 operand fragment from an LDS tile laid out [row][LDK]
// row-major, starting at k offset k_base. CDNA5 ISA 7.12.2 16-bit layout:
//   lane l: row = row_base + (l & 15); hi = l >> 4
//   element i: k = k_base + ((i >> 3) << 4) + hi*8 + (i & 7)
// Element pairs are K-contiguous -> 8x b32 LDS loads (coalesce to b128).
template <int LDK>
__device__ __forceinline__ v16bf load_frag(const unsigned short* lds, int row_base,
                                           int k_base, int lane) {
  Frag16 f;
  const int hi8 = (lane >> 4) << 3;
  const unsigned* p =
      (const unsigned*)(lds + (row_base + (lane & 15)) * LDK + k_base);
#pragma unroll
  for (int j = 0; j < 8; ++j) {
    const int kb = ((j & 4) << 2) + hi8 + ((j & 3) << 1);  // even
    f.u[j] = p[kb >> 1];
  }
  return f.v;
}

// ---------------------------------------------------------------------------
// Tensor Data Mover: 2D bf16 tile (rows x cols elements, row stride ld_elems)
// from global tile-start gptr into LDS at lds_addr. D# per ISA ch.8:
//   group0: count=1 | lds_addr | global_addr[56:0] | type=2
//   group1: data_size=2B, tensor_dim0=ld, tensor_dim1=nrows,
//           tile_dim0=cols, tile_dim1=rows, tensor_dim0_stride=ld
//   groups 2/3 (+extra group of the 6-arg builtin): zero (2D tensor)
// Tile rows pack contiguously into LDS -> matches [row][cols] layout.
// ---------------------------------------------------------------------------
__device__ __forceinline__ void tdm_load_2d_bf16(
    unsigned lds_addr, const void* gptr, unsigned rows, unsigned cols,
    unsigned ld_elems, unsigned tensor_rows) {
  const unsigned long long ga = (unsigned long long)(size_t)gptr;
  v4u g0 = {0u, 0u, 0u, 0u};
  v8i g1 = {0, 0, 0, 0, 0, 0, 0, 0};
  const v4i z4 = {0, 0, 0, 0};
  const v8i z8 = {0, 0, 0, 0, 0, 0, 0, 0};
  g0[0] = 1u;                                            // count=1, no gather
  g0[1] = lds_addr;                                      // lds_addr[31:0]
  g0[2] = (unsigned)ga;                                  // global_addr[31:0]
  g0[3] = (unsigned)((ga >> 32) & 0x01FFFFFFu) | (2u << 30);  // addr[56:32]|type=2
  const unsigned d0 = ld_elems, d1 = tensor_rows;
  g1[0] = (int)(1u << 16);                               // data_size=1 -> 2 bytes
  g1[1] = (int)((d0 & 0xFFFFu) << 16);                   // tensor_dim0[15:0]
  g1[2] = (int)((d0 >> 16) | ((d1 & 0xFFFFu) << 16));    // d0[31:16] | d1[15:0]
  g1[3] = (int)((d1 >> 16) | (cols << 16));              // d1[31:16] | tile_dim0
  g1[4] = (int)rows;                                     // tile_dim1 (tile_dim2=0)
  g1[5] = (int)ld_elems;                                 // tensor_dim0_stride[31:0]
  g1[6] = 0;                                             // stride[47:32]=0
  g1[7] = 0;
  __builtin_amdgcn_tensor_load_to_lds(g0, g1, z4, z4, z8, 0);
}

// ---------------------------------------------------------------------------
// Pre-pass 1: fp32 -> bf16 (pair-packed u32 stores), for X.
// ---------------------------------------------------------------------------
__global__ __launch_bounds__(256) void cvt_bf16_kernel(
    const float* __restrict__ in, unsigned* __restrict__ out2, size_t n2) {
  size_t i = (size_t)blockIdx.x * 256 + threadIdx.x;
  const size_t stride = (size_t)gridDim.x * 256;
  for (; i < n2; i += stride) {
    const float lo = in[2 * i], hi = in[2 * i + 1];
    out2[i] = (unsigned)f2bf_bits(lo) | ((unsigned)f2bf_bits(hi) << 16);
  }
}

// ---------------------------------------------------------------------------
// Pre-pass 2: W[K][N] fp32 -> WT[N][K] bf16, LDS-tiled 32x32 transpose.
// grid = (N/32, K/32). Coalesced fp32 reads along n, u32-paired writes along k.
// ---------------------------------------------------------------------------
__global__ __launch_bounds__(256) void transpose_cvt_kernel(
    const float* __restrict__ W, unsigned short* __restrict__ WT, int K, int N) {
  __shared__ float tile[32][33];
  const int t = threadIdx.x;
  const int n0 = blockIdx.x * 32, k0 = blockIdx.y * 32;
#pragma unroll
  for (int r = 0; r < 4; ++r) {
    const int idx = r * 256 + t;
    const int kk = idx >> 5, nn = idx & 31;
    tile[kk][nn] = W[(size_t)(k0 + kk) * N + n0 + nn];
  }
  __syncthreads();
  unsigned* WT2 = (unsigned*)WT;
#pragma unroll
  for (int r = 0; r < 2; ++r) {
    const int idx = r * 256 + t;           // 512 u32 pairs: 32 nn x 16 k-pairs
    const int nn = idx >> 4, kp = idx & 15;
    const unsigned lo = f2bf_bits(tile[2 * kp][nn]);
    const unsigned hi = f2bf_bits(tile[2 * kp + 1][nn]);
    WT2[((size_t)(n0 + nn) * K + k0) / 2 + kp] = lo | (hi << 16);
  }
}

// ---------------------------------------------------------------------------
// Kernel: confidence head -> continue mask. 32 tokens x 512 conf dims/block.
// ---------------------------------------------------------------------------
__global__ __launch_bounds__(256) void conf_mask_kernel(
    const float* __restrict__ X, const float* __restrict__ Wc1,
    const float* __restrict__ bc1, const float* __restrict__ Wc2,
    const float* __restrict__ bc2, const int* __restrict__ layer_idx,
    unsigned* __restrict__ mask) {
  __shared__ __attribute__((aligned(16))) unsigned short sA[32 * 32];      // [m][k]
  __shared__ __attribute__((aligned(16))) unsigned short sB[D_CONF * 32];  // [n][k]
  __shared__ float sWc2[D_CONF];
  __shared__ float sConf[32];

  const int t = threadIdx.x;
  const int lane = t & 31, wid = t >> 5;
  const int m0 = blockIdx.x * 32;

  if (*layer_idx < MIN_LAYERS) {
    if (t < 32) mask[m0 + t] = 1u;
    return;
  }

  for (int i = t; i < D_CONF; i += 256) sWc2[i] = Wc2[i];
  if (t < 32) sConf[t] = 0.0f;

  v8f acc[2][4];
#pragma unroll
  for (int i = 0; i < 2; ++i)
#pragma unroll
    for (int j = 0; j < 4; ++j) acc[i][j] = (v8f)(0.0f);

  for (int k0 = 0; k0 < D_MODEL; k0 += 32) {
    __syncthreads();
#pragma unroll
    for (int r = 0; r < 4; ++r) {          // A tile 32x32 fp32->bf16
      const int idx = r * 256 + t;
      const int mm = idx >> 5, kk = idx & 31;
      sA[mm * 32 + kk] = f2bf_bits(X[(size_t)(m0 + mm) * D_MODEL + k0 + kk]);
    }
#pragma unroll
    for (int r = 0; r < 64; ++r) {         // B tile 32x512 -> [n][k]
      const int idx = r * 256 + t;
      const int kk = idx >> 9, nn = idx & 511;
      sB[nn * 32 + kk] = f2bf_bits(Wc1[(size_t)(k0 + kk) * D_CONF + nn]);
    }
    __syncthreads();

    const v16bf a0 = load_frag<32>(sA, 0, 0, lane);
    const v16bf a1 = load_frag<32>(sA, 16, 0, lane);
#pragma unroll
    for (int tn = 0; tn < 4; ++tn) {
      const v16bf b = load_frag<32>(sB, wid * 64 + tn * 16, 0, lane);
      acc[0][tn] = __builtin_amdgcn_wmma_f32_16x16x32_bf16(
          false, a0, false, b, (short)0, acc[0][tn], false, false);
      acc[1][tn] = __builtin_amdgcn_wmma_f32_16x16x32_bf16(
          false, a1, false, b, (short)0, acc[1][tn], false, false);
    }
  }

  const int hi = lane >> 4, nl = lane & 15;
#pragma unroll
  for (int tm = 0; tm < 2; ++tm) {
#pragma unroll
    for (int v = 0; v < 8; ++v) {
      float sum = 0.0f;
#pragma unroll
      for (int tn = 0; tn < 4; ++tn) {
        const int n = wid * 64 + tn * 16 + nl;
        const float h = acc[tm][tn][v] + bc1[n];
        sum += gelu_exact(h) * sWc2[n];
      }
      atomicAdd(&sConf[tm * 16 + hi * 8 + v], sum);   // ds_add_f32
    }
  }
  __syncthreads();
  if (t < 32) {
    const float c = sConf[t] + bc2[0];
    const float conf = 1.0f / (1.0f + expf(-c));
    mask[m0 + t] = (conf < THRESHOLD) ? 1u : 0u;
  }
}

// ---------------------------------------------------------------------------
// Kernel: H = GELU(Xbf @ Wf1T^T + bf1), bf16 out. M=NTOK, K=2048, N=8192.
// Double-buffered TDM staging (BLOCK_K=64 -> 16 WMMAs per barrier pair);
// wave 0 issues tiles for step i+1, then s_wait_tensorcnt 2 (in-order TDM
// completion) so the DMA overlaps the current step's WMMAs.
// ---------------------------------------------------------------------------
__global__ __launch_bounds__(256) void ffn1_kernel(
    const unsigned short* __restrict__ Xbf, const unsigned short* __restrict__ Wf1T,
    const float* __restrict__ bf1, unsigned short* __restrict__ H) {
  __shared__ __attribute__((aligned(16))) unsigned short sA[2][128 * 64];  // [m][k]
  __shared__ __attribute__((aligned(16))) unsigned short sB[2][128 * 64];  // [n][k]

  const int t = threadIdx.x, lane = t & 31, wid = t >> 5;
  const int m0 = blockIdx.x * 128, n0 = blockIdx.y * 128;
  const int wm = (wid >> 2) * 64, wn = (wid & 3) * 32;
  const unsigned ldsA[2] = {(unsigned)(size_t)(void*)&sA[0][0],
                            (unsigned)(size_t)(void*)&sA[1][0]};
  const unsigned ldsB[2] = {(unsigned)(size_t)(void*)&sB[0][0],
                            (unsigned)(size_t)(void*)&sB[1][0]};
  const unsigned short* Abase = Xbf  + (size_t)m0 * D_MODEL;
  const unsigned short* Bbase = Wf1T + (size_t)n0 * D_MODEL;

  v8f acc[4][2];
#pragma unroll
  for (int i = 0; i < 4; ++i)
#pragma unroll
    for (int j = 0; j < 2; ++j) acc[i][j] = (v8f)(0.0f);

  const int ITERS = D_MODEL / 64;
  if (wid == 0) {                          // prologue: stage tile 0
    tdm_load_2d_bf16(ldsA[0], Abase, 128, 64, D_MODEL, NTOK);
    tdm_load_2d_bf16(ldsB[0], Bbase, 128, 64, D_MODEL, D_FF);
  }
  for (int it = 0; it < ITERS; ++it) {
    const int p = it & 1;
    if (wid == 0) {
      if (it + 1 < ITERS) {                // stage tile it+1 into idle buffer
        tdm_load_2d_bf16(ldsA[1 - p], Abase + (it + 1) * 64, 128, 64, D_MODEL, NTOK);
        tdm_load_2d_bf16(ldsB[1 - p], Bbase + (it + 1) * 64, 128, 64, D_MODEL, D_FF);
        __builtin_amdgcn_s_wait_tensorcnt(2);   // buffer p complete (in-order)
      } else {
        __builtin_amdgcn_s_wait_tensorcnt(0);
      }
    }
    __syncthreads();                       // buffer p visible to all waves
#pragma unroll
    for (int ks = 0; ks < 2; ++ks) {
      v16bf a[4], b[2];
#pragma unroll
      for (int i = 0; i < 4; ++i) a[i] = load_frag<64>(sA[p], wm + i * 16, ks * 32, lane);
#pragma unroll
      for (int j = 0; j < 2; ++j) b[j] = load_frag<64>(sB[p], wn + j * 16, ks * 32, lane);
#pragma unroll
      for (int i = 0; i < 4; ++i)
#pragma unroll
        for (int j = 0; j < 2; ++j)
          acc[i][j] = __builtin_amdgcn_wmma_f32_16x16x32_bf16(
              false, a[i], false, b[j], (short)0, acc[i][j], false, false);
    }
    __syncthreads();                       // buffer p consumed -> reusable
  }

  const int hi = lane >> 4, nl = lane & 15;
#pragma unroll
  for (int i = 0; i < 4; ++i)
#pragma unroll
    for (int j = 0; j < 2; ++j) {
      const int n = n0 + wn + j * 16 + nl;
      const float bias = bf1[n];
#pragma unroll
      for (int v = 0; v < 8; ++v) {
        const int m = m0 + wm + i * 16 + hi * 8 + v;
        H[(size_t)m * D_FF + n] = f2bf_bits(gelu_exact(acc[i][j][v] + bias));
      }
    }
}

// ---------------------------------------------------------------------------
// Kernel: out = mask ? (H @ Wf2T^T + bf2) : X. M=NTOK, K=8192, N=2048.
// ---------------------------------------------------------------------------
__global__ __launch_bounds__(256) void ffn2_kernel(
    const unsigned short* __restrict__ H, const unsigned short* __restrict__ Wf2T,
    const float* __restrict__ bf2, const float* __restrict__ X,
    const unsigned* __restrict__ mask, float* __restrict__ out) {
  __shared__ __attribute__((aligned(16))) unsigned short sA[2][128 * 64];  // [m][k]
  __shared__ __attribute__((aligned(16))) unsigned short sB[2][128 * 64];  // [n][k]

  const int t = threadIdx.x, lane = t & 31, wid = t >> 5;
  const int m0 = blockIdx.x * 128, n0 = blockIdx.y * 128;
  const int wm = (wid >> 2) * 64, wn = (wid & 3) * 32;
  const unsigned ldsA[2] = {(unsigned)(size_t)(void*)&sA[0][0],
                            (unsigned)(size_t)(void*)&sA[1][0]};
  const unsigned ldsB[2] = {(unsigned)(size_t)(void*)&sB[0][0],
                            (unsigned)(size_t)(void*)&sB[1][0]};
  const unsigned short* Abase = H    + (size_t)m0 * D_FF;
  const unsigned short* Bbase = Wf2T + (size_t)n0 * D_FF;

  v8f acc[4][2];
#pragma unroll
  for (int i = 0; i < 4; ++i)
#pragma unroll
    for (int j = 0; j < 2; ++j) acc[i][j] = (v8f)(0.0f);

  const int ITERS = D_FF / 64;
  if (wid == 0) {
    tdm_load_2d_bf16(ldsA[0], Abase, 128, 64, D_FF, NTOK);
    tdm_load_2d_bf16(ldsB[0], Bbase, 128, 64, D_FF, D_MODEL);
  }
  for (int it = 0; it < ITERS; ++it) {
    const int p = it & 1;
    if (wid == 0) {
      if (it + 1 < ITERS) {
        tdm_load_2d_bf16(ldsA[1 - p], Abase + (it + 1) * 64, 128, 64, D_FF, NTOK);
        tdm_load_2d_bf16(ldsB[1 - p], Bbase + (it + 1) * 64, 128, 64, D_FF, D_MODEL);
        __builtin_amdgcn_s_wait_tensorcnt(2);
      } else {
        __builtin_amdgcn_s_wait_tensorcnt(0);
      }
    }
    __syncthreads();
#pragma unroll
    for (int ks = 0; ks < 2; ++ks) {
      v16bf a[4], b[2];
#pragma unroll
      for (int i = 0; i < 4; ++i) a[i] = load_frag<64>(sA[p], wm + i * 16, ks * 32, lane);
#pragma unroll
      for (int j = 0; j < 2; ++j) b[j] = load_frag<64>(sB[p], wn + j * 16, ks * 32, lane);
#pragma unroll
      for (int i = 0; i < 4; ++i)
#pragma unroll
        for (int j = 0; j < 2; ++j)
          acc[i][j] = __builtin_amdgcn_wmma_f32_16x16x32_bf16(
              false, a[i], false, b[j], (short)0, acc[i][j], false, false);
    }
    __syncthreads();
  }

  const int hi = lane >> 4, nl = lane & 15;
#pragma unroll
  for (int i = 0; i < 4; ++i) {
#pragma unroll
    for (int v = 0; v < 8; ++v) {
      const int m = m0 + wm + i * 16 + hi * 8 + v;
      const unsigned keep = mask[m];
#pragma unroll
      for (int j = 0; j < 2; ++j) {
        const int n = n0 + wn + j * 16 + nl;
        const size_t off = (size_t)m * D_MODEL + n;
        const float val = acc[i][j][v] + bf2[n];
        out[off] = keep ? val : X[off];
      }
    }
  }
}

// ---------------------------------------------------------------------------
extern "C" void kernel_launch(void* const* d_in, const int* in_sizes, int n_in,
                              void* d_out, int out_size, void* d_ws, size_t ws_size,
                              hipStream_t stream) {
  (void)in_sizes; (void)n_in; (void)out_size; (void)ws_size;
  const float* X   = (const float*)d_in[0];
  const float* Wc1 = (const float*)d_in[1];
  const float* bc1 = (const float*)d_in[2];
  const float* Wc2 = (const float*)d_in[3];
  const float* bc2 = (const float*)d_in[4];
  const float* Wf1 = (const float*)d_in[5];
  const float* bf1 = (const float*)d_in[6];
  const float* Wf2 = (const float*)d_in[7];
  const float* bf2 = (const float*)d_in[8];
  const int*   lix = (const int*)d_in[9];
  float* out = (float*)d_out;

  char* ws = (char*)d_ws;
  unsigned short* H    = (unsigned short*)(ws);
  unsigned short* Xbf  = (unsigned short*)(ws + (size_t)NTOK * D_FF * 2);
  unsigned short* Wf1T = (unsigned short*)(ws + (size_t)NTOK * D_FF * 2
                                              + (size_t)NTOK * D_MODEL * 2);
  unsigned short* Wf2T = (unsigned short*)(ws + (size_t)NTOK * D_FF * 2
                                              + (size_t)NTOK * D_MODEL * 2
                                              + (size_t)D_FF * D_MODEL * 2);
  unsigned* mask = (unsigned*)(ws + (size_t)NTOK * D_FF * 2
                                  + (size_t)NTOK * D_MODEL * 2
                                  + (size_t)D_FF * D_MODEL * 4);

  // Pre-pass: one-time bf16 conversion (+ weight transpose to [n][k]).
  cvt_bf16_kernel<<<8192, 256, 0, stream>>>(X, (unsigned*)Xbf,
                                            (size_t)NTOK * D_MODEL / 2);
  transpose_cvt_kernel<<<dim3(D_FF / 32, D_MODEL / 32), 256, 0, stream>>>(
      Wf1, Wf1T, D_MODEL, D_FF);
  transpose_cvt_kernel<<<dim3(D_MODEL / 32, D_FF / 32), 256, 0, stream>>>(
      Wf2, Wf2T, D_FF, D_MODEL);

  conf_mask_kernel<<<NTOK / 32, 256, 0, stream>>>(X, Wc1, bc1, Wc2, bc2, lix, mask);
  ffn1_kernel<<<dim3(NTOK / 128, D_FF / 128), 256, 0, stream>>>(Xbf, Wf1T, bf1, H);
  ffn2_kernel<<<dim3(NTOK / 128, D_MODEL / 128), 256, 0, stream>>>(H, Wf2T, bf2, X, mask, out);
}
